// BasicBlock_63874753626191
// MI455X (gfx1250) — matrix operand order repaired and checked by reading, not compiled, not run on hardware
//
#include <hip/hip_runtime.h>

typedef __attribute__((ext_vector_type(16))) _Float16 v16h;
typedef __attribute__((ext_vector_type(8)))  _Float16 v8h;
typedef __attribute__((ext_vector_type(8)))  float    v8f;
typedef __attribute__((ext_vector_type(4)))  unsigned int v4u;
typedef __attribute__((ext_vector_type(8)))  unsigned int v8u;

#define WMMA_F32_F16(a, b, c) \
  __builtin_amdgcn_wmma_f32_16x16x32_f16(false, (a), false, (b), (short)0, (c), false, false)

constexpr int B_ = 64, T_ = 512, F_ = 16, C_ = 256, H_ = 96;

// ---------------------------------------------------------------------------
// Weight prepack: f32 (Cout, Cin, K) -> f16 [k_shift][Cout][CinPad], where
// tap with time-shift -k*d corresponds to original tap index (K-1-k)
// (causal conv, left pad (K-1)*d). Also used for whh (K=1).
// ---------------------------------------------------------------------------
__global__ void pack_w(const float* __restrict__ w, _Float16* __restrict__ o,
                       int Cout, int Cin, int K, int CinPad) {
  int idx = blockIdx.x * blockDim.x + threadIdx.x;
  int total = K * Cout * CinPad;
  if (idx >= total) return;
  int ci = idx % CinPad;
  int co = (idx / CinPad) % Cout;
  int k  = idx / (CinPad * Cout);
  float v = 0.f;
  if (ci < Cin) v = w[((size_t)co * Cin + ci) * K + (K - 1 - k)];
  o[idx] = (_Float16)v;
}

// inputs (B,T,F) f32 -> X0 f16 channel-last (B, T, 32) zero-padded channels;
// also writes input_main output region (B,T,1).
__global__ void pack_input(const float* __restrict__ inp, _Float16* __restrict__ x0,
                           float* __restrict__ im) {
  int idx = blockIdx.x * blockDim.x + threadIdx.x;
  if (idx >= B_ * T_) return;
  const float* row = inp + (size_t)idx * F_;
  im[idx] = row[0];
  _Float16* orow = x0 + (size_t)idx * 32;
#pragma unroll
  for (int f = 0; f < 32; ++f)
    orow[f] = (f < F_) ? (_Float16)row[f] : (_Float16)0.f;
}

// ---------------------------------------------------------------------------
// Dilated causal conv as implicit GEMM via WMMA, channel-last activations.
//   out[b,t,co] = act( bias[co] + res[b,t,co] + sum_{k,ci} W_k[co,ci]*in[b,t-k*d,ci] )
// Workgroup: 256 threads (8 waves), one 64(t) x 64(co) tile of one batch.
// The 80-row (16-row causal apron) x Cin activation tile is DMA'd into LDS by
// the Tensor Data Mover once per workgroup and reused across all taps and all
// ci chunks; TDM row padding (+4 dwords) staggers LDS rows for bank spread.
// ---------------------------------------------------------------------------
__global__ __launch_bounds__(256) void conv_wmma(
    const _Float16* __restrict__ in,   // (B, T, Cin) f16
    const _Float16* __restrict__ wpk,  // (K, Cout, Cin) f16, tap-shift order
    const float* __restrict__ bias,    // (Cout)
    const _Float16* __restrict__ res,  // (B, T, Cout) f16 or nullptr
    _Float16* __restrict__ out,        // (B, T, Cout) f16
    int Cin, int Cout, int K, int dil, int do_relu) {
  __shared__ _Float16 xT[80 * 264];   // 80 rows x (Cin+8) halfs, max Cin=256
  const int ldsb = Cin + 8;

  const int b    = blockIdx.z;
  const int co0  = blockIdx.y * 64;
  const int t0   = blockIdx.x * 64;
  const int tid  = threadIdx.x;
  const int lane = tid & 31;
  const int wid  = tid >> 5;
  const int coSub = wid >> 1;       // 0..3 : 16-row co sub-tile
  const int tSub0 = (wid & 1) * 2;  // 0/2  : first of two 16-col t sub-tiles
  const int n    = lane & 15;
  const int half = lane >> 4;

  // Pre-zero causal apron rows (t<0) for the leftmost tile (disjoint from TDM rows).
  if (t0 == 0) {
    for (int i = tid; i < 16 * ldsb; i += 256) xT[i] = (_Float16)0.f;
  }
  // One wave issues the TDM descriptor: 2D tile, nRows x Cin halfs, row stride Cin.
  if (wid == 0) {
    int tStart = t0 - 16, ldsRow = 0;
    if (tStart < 0) { ldsRow = -tStart; tStart = 0; }
    int nRows = 80 - ldsRow;
    unsigned int ldsAddr = (unsigned int)(uintptr_t)(&xT[0]) +
                           (unsigned int)(ldsRow * ldsb * 2);
    unsigned long long ga =
        (unsigned long long)(uintptr_t)(in + ((size_t)b * T_ + tStart) * Cin);
    unsigned int padInt = (unsigned int)(__builtin_ctz((unsigned)Cin) - 2); // Cin/2 dwords
    v4u g0;
    g0[0] = 1u;                                  // count=1, user descriptor
    g0[1] = ldsAddr;                             // lds_addr (bytes)
    g0[2] = (unsigned int)ga;                    // global_addr[31:0]
    g0[3] = ((unsigned int)(ga >> 32) & 0x01FFFFFFu) | (2u << 30); // addr[56:32], type=2
    v8u g1;
    g1[0] = (1u << 16)                           // data_size = 2 bytes
          | (1u << 20)                           // pad_enable
          | (padInt << 22)                       // pad_interval = Cin/2 dwords
          | (3u << 25);                          // pad_amount = 4 dwords (8 halfs)
    g1[1] = ((unsigned int)Cin & 0xFFFFu) << 16;                 // tensor_dim0 lo16
    g1[2] = ((unsigned int)Cin >> 16) | (((unsigned int)nRows & 0xFFFFu) << 16);
    g1[3] = ((unsigned int)nRows >> 16) | ((unsigned int)Cin << 16); // tile_dim0 = Cin
    g1[4] = (unsigned int)nRows;                 // tile_dim1 = nRows, tile_dim2 = 0
    g1[5] = (unsigned int)Cin;                   // tensor_dim0_stride[31:0]
    g1[6] = 0u;
    g1[7] = 0u;
    v4u gz = (v4u){0u, 0u, 0u, 0u};              // groups 2/3: unused dims
    asm volatile("tensor_load_to_lds %0, %1, %2, %3"
                 :: "s"(g0), "s"(g1), "s"(gz), "s"(gz)
                 : "memory");
    __builtin_amdgcn_s_wait_tensorcnt(0);
  }
  __syncthreads();

  v8f acc0 = {}; v8f acc1 = {};
  for (int k = 0; k < K; ++k) {
    const _Float16* Wk = wpk + (size_t)k * Cout * Cin;
    const int rbase = 16 - k * dil;   // LDS row of global time t0 - k*dil
    for (int kc = 0; kc < Cin; kc += 32) {
      // A fragment: weight rows co0+coSub*16..+15, K-slice [kc, kc+32)
      const _Float16* ap = Wk + (size_t)(co0 + coSub * 16 + n) * Cin + kc + half * 16;
      __builtin_prefetch(ap + 32, 0, 1);
      v8h alo = *(const v8h*)ap;
      v8h ahi = *(const v8h*)(ap + 8);
      v16h a;
#pragma unroll
      for (int i = 0; i < 8; ++i) { a[i] = alo[i]; a[i + 8] = ahi[i]; }
      // B fragments: contiguous 32B per lane from the TDM-staged tile
      const _Float16* bp0 = &xT[(rbase + tSub0 * 16 + n) * ldsb + kc + half * 16];
      v16h bb0 = *(const v16h*)bp0;
      acc0 = WMMA_F32_F16(a, bb0, acc0);
      const _Float16* bp1 = &xT[(rbase + (tSub0 + 1) * 16 + n) * ldsb + kc + half * 16];
      v16h bb1 = *(const v16h*)bp1;
      acc1 = WMMA_F32_F16(a, bb1, acc1);
    }
  }

  // Epilogue: bias + optional residual + relu; one packed 16B store per lane/tile.
  const int coB = co0 + coSub * 16 + half * 8;
  {
    int t = t0 + tSub0 * 16 + n;
    size_t base = ((size_t)b * T_ + t) * Cout + coB;
    v8h rv = {};
    if (res) rv = *(const v8h*)(res + base);
    v8h o;
#pragma unroll
    for (int r = 0; r < 8; ++r) {
      float v = acc0[r] + bias[coB + r];
      if (res) v += (float)rv[r];
      if (do_relu) v = fmaxf(v, 0.f);
      o[r] = (_Float16)v;
    }
    *(v8h*)(out + base) = o;
  }
  {
    int t = t0 + (tSub0 + 1) * 16 + n;
    size_t base = ((size_t)b * T_ + t) * Cout + coB;
    v8h rv = {};
    if (res) rv = *(const v8h*)(res + base);
    v8h o;
#pragma unroll
    for (int r = 0; r < 8; ++r) {
      float v = acc1[r] + bias[coB + r];
      if (res) v += (float)rv[r];
      if (do_relu) v = fmaxf(v, 0.f);
      o[r] = (_Float16)v;
    }
    *(v8h*)(out + base) = o;
  }
}

__global__ void extract_hidden(const _Float16* __restrict__ act, float* __restrict__ hid) {
  int idx = blockIdx.x * blockDim.x + threadIdx.x;
  if (idx >= B_ * C_) return;
  int b = idx >> 8, c = idx & 255;
  hid[idx] = (float)act[((size_t)b * T_ + (T_ - 1)) * C_ + c];
}

// ---------------------------------------------------------------------------
// Persistent LSTM decoder: one 1024-thread workgroup (32 waves) runs the whole
// sequential scan. Per step: g = h @ whh.T (WMMA, A from LDS h, B from f16 whh)
// + x*wih + bias -> LDS (256KB f32), then cell update (c in registers, 16
// cells/thread), h back to LDS as f16, y = h.fcw + fcb via LDS float atomics.
// ---------------------------------------------------------------------------
__global__ __launch_bounds__(1024) void lstm_wmma(
    const float* __restrict__ hid,     // (64,256) h0
    const float* __restrict__ x0v,     // (64)
    const _Float16* __restrict__ whh,  // (1024,256) f16 row-major
    const float* __restrict__ wih,     // (1024)
    const float* __restrict__ bih,     // (1024)
    const float* __restrict__ bhh,     // (1024)
    const float* __restrict__ fcw,     // (256)
    const float* __restrict__ fcb,     // (1)
    float* __restrict__ yout,          // (64, steps)
    int steps) {
  constexpr int HP = 264;  // padded h row (halfs)
  extern __shared__ char smem[];
  _Float16* hbuf = (_Float16*)smem;                       // 64*HP halfs
  float* gbuf = (float*)(smem + 64 * HP * 2);             // 64*1024 f32
  float* xbuf = (float*)((char*)gbuf + 64 * 1024 * 4);    // 64
  float* ybuf = xbuf + 64;                                // 64

  const int tid  = threadIdx.x;
  const int lane = tid & 31;
  const int wid  = tid >> 5;
  const int n    = lane & 15;
  const int half = lane >> 4;

  for (int i = tid; i < B_ * C_; i += 1024) {
    int b = i >> 8, c = i & 255;
    hbuf[b * HP + c] = (_Float16)hid[i];
  }
  if (tid < 64) { xbuf[tid] = x0v[tid]; ybuf[tid] = 0.f; }

  float creg[16];
#pragma unroll
  for (int q = 0; q < 16; ++q) creg[q] = 0.f;

  const int mt = wid >> 3;           // 0..3 batch tile
  const int ntBase = (wid & 7) * 8;  // 8 consecutive N-tiles per wave
  const int b0 = mt * 16;
  const int cb = tid >> 4;           // batch for cell phase
  const int j0 = (tid & 15) * 16;

  __syncthreads();

  for (int s = 0; s < steps; ++s) {
    // ---- gate GEMM: g = h @ whh.T ----
    v8f acc[8];
#pragma unroll
    for (int i = 0; i < 8; ++i) acc[i] = (v8f){};
#pragma unroll
    for (int kc = 0; kc < 256; kc += 32) {
      const _Float16* hp = &hbuf[(b0 + n) * HP + kc + half * 16];
      v8h lo = *(const v8h*)hp;
      v8h hi = *(const v8h*)(hp + 8);
      v16h a;
#pragma unroll
      for (int i = 0; i < 8; ++i) { a[i] = lo[i]; a[i + 8] = hi[i]; }
#pragma unroll
      for (int i = 0; i < 8; ++i) {
        int n0 = (ntBase + i) * 16;
        const _Float16* bp = whh + (size_t)(n0 + n) * 256 + kc + half * 16;
        v16h bb = *(const v16h*)bp;
        acc[i] = WMMA_F32_F16(a, bb, acc[i]);
      }
    }
#pragma unroll
    for (int i = 0; i < 8; ++i) {
      int j = (ntBase + i) * 16 + n;
      float wj = wih[j];
      float bj = bih[j] + bhh[j];
#pragma unroll
      for (int r = 0; r < 8; ++r) {
        int b = b0 + half * 8 + r;
        gbuf[b * 1024 + j] = acc[i][r] + xbuf[b] * wj + bj;
      }
    }
    __syncthreads();

    // ---- cell update (16 cells / thread, same batch) ----
    const float* g = &gbuf[cb * 1024];
    float part = 0.f;
#pragma unroll
    for (int q = 0; q < 16; ++q) {
      int j = j0 + q;
      float ig = 1.f / (1.f + __expf(-g[j]));
      float fg = 1.f / (1.f + __expf(-g[256 + j]));
      float gg = tanhf(g[512 + j]);
      float og = 1.f / (1.f + __expf(-g[768 + j]));
      float c = fg * creg[q] + ig * gg;
      creg[q] = c;
      float h = og * tanhf(c);
      hbuf[cb * HP + j] = (_Float16)h;
      part += h * fcw[j];
    }
    atomicAdd(&ybuf[cb], part);
    __syncthreads();

    // ---- y, next x ----
    if (tid < 64) {
      float y = ybuf[tid] + fcb[0];
      yout[(size_t)tid * steps + s] = y;
      xbuf[tid] = y;
      ybuf[tid] = 0.f;
    }
    __syncthreads();
  }
}

// ---------------------------------------------------------------------------
// Small dense heads (VALU)
// ---------------------------------------------------------------------------
__global__ void ar_head(const float* __restrict__ hid, const float* __restrict__ w,
                        const float* __restrict__ bias, float* __restrict__ outAr,
                        float* __restrict__ fx0) {
  int idx = blockIdx.x * blockDim.x + threadIdx.x;
  if (idx >= B_ * H_) return;
  int b = idx / H_, h = idx % H_;
  float s = bias[h];
  const float* hr = hid + (size_t)b * C_;
  const float* wr = w + (size_t)h * C_;
  for (int c = 0; c < C_; ++c) s += hr[c] * wr[c];
  outAr[idx] = s;
  if (h == 0) fx0[b] = s;
}

__global__ void h2i_head(const float* __restrict__ hid, const float* __restrict__ w,
                         const float* __restrict__ bias, float* __restrict__ ex0) {
  int b = blockIdx.x * blockDim.x + threadIdx.x;
  if (b >= B_) return;
  float s = bias[0];
  for (int c = 0; c < C_; ++c) s += hid[b * C_ + c] * w[c];
  ex0[b] = s;
}

__global__ void dfb_kernel(const float* __restrict__ im, const float* __restrict__ est,
                           float* __restrict__ a) {
  int idx = blockIdx.x * blockDim.x + threadIdx.x;
  if (idx >= B_ * T_) return;
  a[idx] = im[idx] - est[idx];
}

__global__ void dag1(const float* __restrict__ a, const float* __restrict__ w1,
                     const float* __restrict__ b1, float* __restrict__ t1) {
  int idx = blockIdx.x * blockDim.x + threadIdx.x;
  if (idx >= B_ * T_) return;
  int b = idx / T_, u = idx % T_;
  float s = b1[u];
  const float* ar = a + (size_t)b * T_;
  const float* wr = w1 + (size_t)u * T_;
  for (int t = 0; t < T_; ++t) s += ar[t] * wr[t];
  t1[idx] = fmaxf(s, 0.f);
}

__global__ void dag2(const float* __restrict__ t1, const float* __restrict__ w2,
                     const float* __restrict__ b2, const float* __restrict__ a,
                     const float* __restrict__ im, float* __restrict__ nextin,
                     float* __restrict__ gobs) {
  int idx = blockIdx.x * blockDim.x + threadIdx.x;
  if (idx >= B_ * T_) return;
  int b = idx / T_, t2 = idx % T_;
  float s = b2[t2];
  const float* tr = t1 + (size_t)b * T_;
  const float* wr = w2 + (size_t)t2 * T_;
  for (int u = 0; u < T_; ++u) s += tr[u] * wr[u];
  float alpha = 1.f / (1.f + __expf(-s));
  float gv = alpha * a[idx];
  gobs[idx] = gv;
  nextin[idx] = im[idx] - gv;
}

__global__ void gds_head(const float* __restrict__ gobs, const float* __restrict__ w,
                         const float* __restrict__ bias, float* __restrict__ outg) {
  int idx = blockIdx.x * blockDim.x + threadIdx.x;
  if (idx >= B_ * H_) return;
  int b = idx / H_, h = idx % H_;
  float s = bias[h];
  const float* gr = gobs + (size_t)b * T_;
  const float* wr = w + (size_t)h * T_;
  for (int t = 0; t < T_; ++t) s += gr[t] * wr[t];
  outg[idx] = s;
}

// ---------------------------------------------------------------------------
extern "C" void kernel_launch(void* const* d_in, const int* in_sizes, int n_in,
                              void* d_out, int out_size, void* d_ws, size_t ws_size,
                              hipStream_t stream) {
  (void)in_sizes; (void)n_in; (void)out_size; (void)ws_size;
  const float* inputs = (const float*)d_in[0];
  const float* l0_w1 = (const float*)d_in[1];  const float* l0_b1 = (const float*)d_in[2];
  const float* l0_w2 = (const float*)d_in[3];  const float* l0_b2 = (const float*)d_in[4];
  const float* l0_dw = (const float*)d_in[5];  const float* l0_db = (const float*)d_in[6];
  const float* l1_w1 = (const float*)d_in[7];  const float* l1_b1 = (const float*)d_in[8];
  const float* l1_w2 = (const float*)d_in[9];  const float* l1_b2 = (const float*)d_in[10];
  const float* l2_w1 = (const float*)d_in[11]; const float* l2_b1 = (const float*)d_in[12];
  const float* l2_w2 = (const float*)d_in[13]; const float* l2_b2 = (const float*)d_in[14];
  const float* ar_w  = (const float*)d_in[15]; const float* ar_b  = (const float*)d_in[16];
  const float* h2i_w = (const float*)d_in[17]; const float* h2i_b = (const float*)d_in[18];
  const float* f_wih = (const float*)d_in[19]; const float* f_whh = (const float*)d_in[20];
  const float* f_bih = (const float*)d_in[21]; const float* f_bhh = (const float*)d_in[22];
  const float* f_fcw = (const float*)d_in[23]; const float* f_fcb = (const float*)d_in[24];
  const float* e_wih = (const float*)d_in[25]; const float* e_whh = (const float*)d_in[26];
  const float* e_bih = (const float*)d_in[27]; const float* e_bhh = (const float*)d_in[28];
  const float* e_fcw = (const float*)d_in[29]; const float* e_fcb = (const float*)d_in[30];
  const float* dag_w1 = (const float*)d_in[31]; const float* dag_b1 = (const float*)d_in[32];
  const float* dag_w2 = (const float*)d_in[33]; const float* dag_b2 = (const float*)d_in[34];
  const float* gds_w  = (const float*)d_in[35]; const float* gds_b  = (const float*)d_in[36];

  // output regions (concatenated tuple, return order)
  float* out  = (float*)d_out;
  float* oF   = out;                  // forecast_outputs (64,96)
  float* oNI  = out + 6144;           // next_input       (64,512)
  float* oAR  = out + 38912;          // ar_predictions   (64,96)
  float* oEST = out + 45056;          // estimate_outputs (64,512)
  float* oIM  = out + 77824;          // input_main       (64,512)
  float* oGH  = out + 110592;         // gds_horizon      (64,96)

  // workspace carve
  char* ws = (char*)d_ws;
  size_t off = 0;
  auto carve = [&](size_t bytes) -> char* {
    char* p = ws + off;
    off += (bytes + 255) & ~(size_t)255;
    return p;
  };
  const size_t actBytes = (size_t)B_ * C_ * T_ * 2;
  _Float16* X0    = (_Float16*)carve((size_t)B_ * 32 * T_ * 2);
  _Float16* ACT_A = (_Float16*)carve(actBytes);
  _Float16* ACT_B = (_Float16*)carve(actBytes);
  _Float16* ACT_C = (_Float16*)carve(actBytes);
  _Float16* Wl0w1 = (_Float16*)carve(4 * 256 * 32 * 2);
  _Float16* Wl0dw = (_Float16*)carve(1 * 256 * 32 * 2);
  _Float16* Wl0w2 = (_Float16*)carve(4 * 256 * 256 * 2);
  _Float16* Wl1w1 = (_Float16*)carve(4 * 256 * 256 * 2);
  _Float16* Wl1w2 = (_Float16*)carve(4 * 256 * 256 * 2);
  _Float16* Wl2w1 = (_Float16*)carve(4 * 256 * 256 * 2);
  _Float16* Wl2w2 = (_Float16*)carve(4 * 256 * 256 * 2);
  _Float16* WHHF  = (_Float16*)carve(1024 * 256 * 2);
  _Float16* WHHE  = (_Float16*)carve(1024 * 256 * 2);
  float* HID  = (float*)carve((size_t)B_ * C_ * 4);
  float* FX0  = (float*)carve(B_ * 4);
  float* EX0  = (float*)carve(B_ * 4);
  float* ADFB = (float*)carve((size_t)B_ * T_ * 4);
  float* T1B  = (float*)carve((size_t)B_ * T_ * 4);
  float* GOBS = (float*)carve((size_t)B_ * T_ * 4);

  // ---- pack inputs & weights ----
  pack_input<<<(B_ * T_ + 255) / 256, 256, 0, stream>>>(inputs, X0, oIM);
  auto launch_pack = [&](const float* w, _Float16* o, int Cout, int Cin, int K, int CinPad) {
    int total = K * Cout * CinPad;
    pack_w<<<(total + 255) / 256, 256, 0, stream>>>(w, o, Cout, Cin, K, CinPad);
  };
  launch_pack(l0_w1, Wl0w1, 256, 16, 4, 32);
  launch_pack(l0_dw, Wl0dw, 256, 16, 1, 32);
  launch_pack(l0_w2, Wl0w2, 256, 256, 4, 256);
  launch_pack(l1_w1, Wl1w1, 256, 256, 4, 256);
  launch_pack(l1_w2, Wl1w2, 256, 256, 4, 256);
  launch_pack(l2_w1, Wl2w1, 256, 256, 4, 256);
  launch_pack(l2_w2, Wl2w2, 256, 256, 4, 256);
  launch_pack(f_whh, WHHF, 1024, 256, 1, 256);
  launch_pack(e_whh, WHHE, 1024, 256, 1, 256);

  // ---- TCN stack (WMMA implicit GEMMs, TDM-staged tiles) ----
  dim3 cg(T_ / 64, C_ / 64, B_);
  // block0: o1 = relu(conv(X0,w1)); res = conv(X0,dw)+db; out = relu(conv(o1,w2)+b2+res)
  conv_wmma<<<cg, 256, 0, stream>>>(X0, Wl0w1, l0_b1, nullptr, ACT_A, 32, 256, 4, 1, 1);
  conv_wmma<<<cg, 256, 0, stream>>>(X0, Wl0dw, l0_db, nullptr, ACT_C, 32, 256, 1, 1, 0);
  conv_wmma<<<cg, 256, 0, stream>>>(ACT_A, Wl0w2, l0_b2, ACT_C, ACT_B, 256, 256, 4, 1, 1);
  // block1 (d=2), identity residual
  conv_wmma<<<cg, 256, 0, stream>>>(ACT_B, Wl1w1, l1_b1, nullptr, ACT_A, 256, 256, 4, 2, 1);
  conv_wmma<<<cg, 256, 0, stream>>>(ACT_A, Wl1w2, l1_b2, ACT_B, ACT_C, 256, 256, 4, 2, 1);
  // block2 (d=4), identity residual
  conv_wmma<<<cg, 256, 0, stream>>>(ACT_C, Wl2w1, l2_b1, nullptr, ACT_A, 256, 256, 4, 4, 1);
  conv_wmma<<<cg, 256, 0, stream>>>(ACT_A, Wl2w2, l2_b2, ACT_C, ACT_B, 256, 256, 4, 4, 1);

  // ---- heads / LSTMs ----
  extract_hidden<<<(B_ * C_ + 255) / 256, 256, 0, stream>>>(ACT_B, HID);
  ar_head<<<(B_ * H_ + 255) / 256, 256, 0, stream>>>(HID, ar_w, ar_b, oAR, FX0);
  h2i_head<<<1, 64, 0, stream>>>(HID, h2i_w, h2i_b, EX0);

  size_t ldsBytes = (size_t)64 * 264 * 2 + (size_t)64 * 1024 * 4 + 64 * 4 + 64 * 4;
  lstm_wmma<<<1, 1024, ldsBytes, stream>>>(HID, FX0, WHHF, f_wih, f_bih, f_bhh,
                                           f_fcw, f_fcb, oF, 96);
  lstm_wmma<<<1, 1024, ldsBytes, stream>>>(HID, EX0, WHHE, e_wih, e_bih, e_bhh,
                                           e_fcw, e_fcb, oEST, 512);

  // ---- DAG / gds ----
  dfb_kernel<<<(B_ * T_ + 255) / 256, 256, 0, stream>>>(oIM, oEST, ADFB);
  dag1<<<(B_ * T_ + 127) / 128, 128, 0, stream>>>(ADFB, dag_w1, dag_b1, T1B);
  dag2<<<(B_ * T_ + 127) / 128, 128, 0, stream>>>(T1B, dag_w2, dag_b2, ADFB, oIM, oNI, GOBS);
  gds_head<<<(B_ * H_ + 255) / 256, 256, 0, stream>>>(GOBS, gds_w, gds_b, oGH);
}
